// XConv_32126355374212
// MI455X (gfx1250) — compile-verified, matrix operand order checked by reading, not run on hardware
//
#include <hip/hip_runtime.h>

#define K_NB   8
#define C_IN   64
#define C_OUT  128
#define TILE_M 16
#define KDIM   1024   // (C_lift + C_in) * K_NB, flattened j' = c*8 + k
#define HALF_J 512

typedef _Float16 half8  __attribute__((ext_vector_type(8)));
typedef _Float16 half16 __attribute__((ext_vector_type(16)));
typedef float    float8 __attribute__((ext_vector_type(8)));
typedef float    float4v __attribute__((ext_vector_type(4)));

struct XConvArgs {
    const float* p;
    const float* P;
    const float* F;
    const float* lW0; const float* lW1; const float* lW2;
    const float* lb0; const float* lb1; const float* lb2;
    const float* mW[49];
    const float* mb[49];
    int mdim[50];
    const _Float16* Bt;   // f16 cast of Kw, (C_OUT, 1024) row-major
    float* out;
};

// ---------------------------------------------------------------- prep: Bt = (f16)Kw
__global__ void cast_kw_kernel(const float* __restrict__ Kw,
                               _Float16* __restrict__ Bt, int nelem) {
    int i = blockIdx.x * blockDim.x + threadIdx.x;
    if (i < nelem) Bt[i] = (_Float16)Kw[i];
}

// ---------------------------------------------------------------- fused XConv
__launch_bounds__(128)
__global__ void xconv_fused_kernel(XConvArgs a) {
    // LDS: 4KB + 16KB + 16KB = 36KB
    __shared__ __align__(16) float     Xs [TILE_M][K_NB][K_NB];   // [m][k-row][t-col]
    __shared__ __align__(16) _Float16  Fc [TILE_M][64][K_NB];     // [m][c][t] current c-half
    __shared__ __align__(16) _Float16  FXh[TILE_M][HALF_J];       // [m][c*8+k] f16 A rows

    const int tid = threadIdx.x;
    const int m   = tid >> 3;          // point within tile
    const int k   = tid & 7;           // neighbor
    const int n   = blockIdx.x * TILE_M + m;

    // ---------- local coordinates
    float pl[3];
#pragma unroll
    for (int d = 0; d < 3; ++d)
        pl[d] = a.P[((size_t)n * K_NB + k) * 3 + d] - a.p[(size_t)n * 3 + d];

    // ---------- X transform MLP: 49 tiny layers (<=8 wide), ReLU each layer
    float xa[8];
    xa[0] = pl[0]; xa[1] = pl[1]; xa[2] = pl[2];
#pragma unroll
    for (int i = 3; i < 8; ++i) xa[i] = 0.f;

    for (int l = 0; l < 49; ++l) {
        const float* W  = a.mW[l];
        const float* Bb = a.mb[l];
        const int ind  = a.mdim[l];
        const int outd = a.mdim[l + 1];
        float xb[8];
#pragma unroll
        for (int o = 0; o < 8; ++o) {
            float acc = 0.f;
            if (o < outd) {
                acc = Bb[o];
#pragma unroll
                for (int i = 0; i < 8; ++i) {
                    float w = W[(i < ind) ? (i * outd + o) : 0];
                    acc += (i < ind) ? xa[i] * w : 0.f;
                }
                acc = fmaxf(acc, 0.f);
            }
            xb[o] = acc;
        }
#pragma unroll
        for (int o = 0; o < 8; ++o) xa[o] = xb[o];
    }
#pragma unroll
    for (int t = 0; t < 8; ++t) Xs[m][k][t] = xa[t];

    // ---------- lift MLP 3->8->23->64 (ReLU each layer) into Fc[m][c][k]
    float h1[8];
#pragma unroll
    for (int o = 0; o < 8; ++o) {
        float acc = a.lb0[o];
#pragma unroll
        for (int i = 0; i < 3; ++i) acc += pl[i] * a.lW0[i * 8 + o];
        h1[o] = fmaxf(acc, 0.f);
    }
    float h2[23];
#pragma unroll
    for (int o = 0; o < 23; ++o) {
        float acc = a.lb1[o];
#pragma unroll
        for (int i = 0; i < 8; ++i) acc += h1[i] * a.lW1[i * 23 + o];
        h2[o] = fmaxf(acc, 0.f);
    }
    for (int o = 0; o < 64; ++o) {
        float acc = a.lb2[o];
#pragma unroll
        for (int i = 0; i < 23; ++i) acc += h2[i] * a.lW2[i * 64 + o];
        Fc[m][o][k] = (_Float16)fmaxf(acc, 0.f);
    }

    // ---------- WMMA state: wave wv owns output col-tiles {2wv, 2wv+1}
    const int wv    = tid >> 5;
    const int ln    = tid & 31;
    const int mrow  = ln & 15;   // A: row m / B: col o / C: col
    const int khalf = ln >> 4;   // K-striping half per ISA 7.12.2
    float8 acc0 = {0.f,0.f,0.f,0.f,0.f,0.f,0.f,0.f};
    float8 acc1 = {0.f,0.f,0.f,0.f,0.f,0.f,0.f,0.f};

    for (int pass = 0; pass < 2; ++pass) {
        if (pass == 1) {
            __syncthreads();  // FXh/Fc reads of pass 0 complete
            // Fc <- raw features F[n, k, 0..63]
            const float4v* Fp = (const float4v*)(a.F + ((size_t)n * K_NB + k) * C_IN);
#pragma unroll
            for (int c4 = 0; c4 < 16; ++c4) {
                float4v v = Fp[c4];
#pragma unroll
                for (int j = 0; j < 4; ++j) Fc[m][c4 * 4 + j][k] = (_Float16)v[j];
            }
        }
        __syncthreads();

        // ---------- F_X half: FXh[m][c*8+k] = sum_t Xs[m][k][t] * Fc[m][c][t]
        {
            float xl[8][8];
#pragma unroll
            for (int kk = 0; kk < 8; ++kk) {
                float4v u = *(const float4v*)&Xs[m][kk][0];
                float4v v = *(const float4v*)&Xs[m][kk][4];
#pragma unroll
                for (int t = 0; t < 4; ++t) { xl[kk][t] = u[t]; xl[kk][4 + t] = v[t]; }
            }
#pragma unroll
            for (int c = 0; c < 8; ++c) {
                const int cl = k * 8 + c;   // this thread's c-group = its k slot
                half8 fch = *(const half8*)&Fc[m][cl][0];
                float fcf[8];
#pragma unroll
                for (int t = 0; t < 8; ++t) fcf[t] = (float)fch[t];
                half8 outh;
#pragma unroll
                for (int kk = 0; kk < 8; ++kk) {
                    float s = 0.f;
#pragma unroll
                    for (int t = 0; t < 8; ++t) s += xl[kk][t] * fcf[t];
                    outh[kk] = (_Float16)s;
                }
                *(half8*)&FXh[m][cl * 8] = outh;
            }
        }
        __syncthreads();

        // ---------- WMMA: A rows = FXh (16 x 512 this pass), B = Bt, 16 k-steps
        const int jpass = pass * HALF_J;
        for (int step = 0; step < 16; ++step) {
            const int jl = step * 32 + khalf * 8;
            union { half16 v; half8 h[2]; } A;
            A.h[0] = *(const half8*)&FXh[mrow][jl];
            A.h[1] = *(const half8*)&FXh[mrow][jl + 16];
            const int jg = jpass + jl;
            {
                const _Float16* bp = a.Bt + (size_t)((wv * 2 + 0) * 16 + mrow) * KDIM + jg;
                union { half16 v; half8 h[2]; } Bv;
                Bv.h[0] = *(const half8*)bp;
                Bv.h[1] = *(const half8*)(bp + 16);
                acc0 = __builtin_amdgcn_wmma_f32_16x16x32_f16(
                    false, A.v, false, Bv.v, (short)0, acc0, false, false);
            }
            {
                const _Float16* bp = a.Bt + (size_t)((wv * 2 + 1) * 16 + mrow) * KDIM + jg;
                union { half16 v; half8 h[2]; } Bv;
                Bv.h[0] = *(const half8*)bp;
                Bv.h[1] = *(const half8*)(bp + 16);
                acc1 = __builtin_amdgcn_wmma_f32_16x16x32_f16(
                    false, A.v, false, Bv.v, (short)0, acc1, false, false);
            }
        }
    }

    // ---------- C epilogue: VGPR r -> row khalf*8 + r, col = tile*16 + mrow
    const size_t rowbase = (size_t)blockIdx.x * TILE_M + khalf * 8;
#pragma unroll
    for (int r = 0; r < 8; ++r) {
        a.out[(rowbase + r) * C_OUT + (wv * 2 + 0) * 16 + mrow] = acc0[r];
        a.out[(rowbase + r) * C_OUT + (wv * 2 + 1) * 16 + mrow] = acc1[r];
    }
}

// ---------------------------------------------------------------- host launch
// d_in order: 0=p, 1=P, 2=F, 3..5=lift_Ws, 6..8=lift_bs,
//             9..57=mlp_Ws(49), 58..106=mlp_bs(49), 107=Kw
extern "C" void kernel_launch(void* const* d_in, const int* in_sizes, int n_in,
                              void* d_out, int out_size, void* d_ws, size_t ws_size,
                              hipStream_t stream) {
    XConvArgs a;
    a.p   = (const float*)d_in[0];
    a.P   = (const float*)d_in[1];
    a.F   = (const float*)d_in[2];
    a.lW0 = (const float*)d_in[3];
    a.lW1 = (const float*)d_in[4];
    a.lW2 = (const float*)d_in[5];
    a.lb0 = (const float*)d_in[6];
    a.lb1 = (const float*)d_in[7];
    a.lb2 = (const float*)d_in[8];
    for (int l = 0; l < 49; ++l) {
        a.mW[l] = (const float*)d_in[9 + l];
        a.mb[l] = (const float*)d_in[58 + l];
    }
    a.mdim[0] = 3;
    for (int l = 0; l < 49; ++l) a.mdim[l + 1] = in_sizes[58 + l];

    _Float16* Bt = (_Float16*)d_ws;   // 128*1024*2 = 256 KB
    a.Bt  = Bt;
    a.out = (float*)d_out;

    const float* Kw = (const float*)d_in[107];
    const int kwn = in_sizes[107];    // 131072
    cast_kw_kernel<<<(kwn + 255) / 256, 256, 0, stream>>>(Kw, Bt, kwn);

    const int N = in_sizes[0] / 3;    // 100000, multiple of 16
    xconv_fused_kernel<<<N / TILE_M, 128, 0, stream>>>(a);
}
